// CompressedLinear_1529008357750
// MI455X (gfx1250) — matrix operand order repaired and checked by reading, not compiled, not run on hardware
//
#include <hip/hip_runtime.h>

// ---------------------------------------------------------------------------
// CompressedLinear: out[M,N] = (x_f16[M,K] @ w_f16[N,K]^T) * scale[N] + bias[N]
//   M = B*S = 8192, N = OUT = 4096, K = IN = 4096
//   w_int8 is exact in f16; scale applied post-accumulation in f32.
//   Block tile 256x128 (8 waves, 4Mx2N), wave tile 64x64 = 4x4 WMMA tiles.
//   Double-buffered LDS filled via GLOBAL_LOAD_ASYNC_TO_LDS (ASYNCcnt path).
// ---------------------------------------------------------------------------

typedef __attribute__((ext_vector_type(16))) _Float16 v16h;
typedef __attribute__((ext_vector_type(8)))  _Float16 v8h;
typedef __attribute__((ext_vector_type(4)))  _Float16 v4h;
typedef __attribute__((ext_vector_type(8)))  float    v8f;

#define OUT_DIM 4096
#define IN_DIM  4096
#define M_DIM   8192            // B * S = 4 * 2048

constexpr int BM = 256;         // block tile M
constexpr int BN = 128;         // block tile N
constexpr int BK = 32;          // K step == WMMA K
constexpr int LDS_STRIDE = 40;  // padded halves/row: 20r mod 64 distinct -> conflict-free

// ---------------------------------------------------------------------------
// CDNA5 async copy: memory -> LDS, tracked by ASYNCcnt (no VGPR staging).
// ---------------------------------------------------------------------------
__device__ __forceinline__ void async_copy_b128(unsigned lds_off, const void* gaddr) {
    asm volatile("global_load_async_to_lds_b128 %0, %1, off"
                 :: "v"(lds_off), "v"(gaddr)
                 : "memory");
}

__device__ __forceinline__ void wait_asynccnt0() {
    asm volatile("s_wait_asynccnt 0" ::: "memory");
}

__device__ __forceinline__ unsigned lds_addr_of(const void* p) {
    // Generic pointers to LDS carry the workgroup-relative LDS offset in [31:0].
    return (unsigned)(unsigned long long)p;
}

// ---------------------------------------------------------------------------
// Conversion kernels: stage f16 operands into workspace (L2-resident afterwards)
// ---------------------------------------------------------------------------
__global__ void convert_w_f16(const int4* __restrict__ w, v4h* __restrict__ o, int n4) {
    int i = blockIdx.x * blockDim.x + threadIdx.x;
    if (i < n4) {
        int4 v = w[i];
        v4h h;
        h[0] = (_Float16)(float)v.x;   // int8 range: exact in f16
        h[1] = (_Float16)(float)v.y;
        h[2] = (_Float16)(float)v.z;
        h[3] = (_Float16)(float)v.w;
        o[i] = h;
    }
}

__global__ void convert_x_f16(const float4* __restrict__ x, v4h* __restrict__ o, int n4) {
    int i = blockIdx.x * blockDim.x + threadIdx.x;
    if (i < n4) {
        float4 v = x[i];
        v4h h;
        h[0] = (_Float16)v.x;
        h[1] = (_Float16)v.y;
        h[2] = (_Float16)v.z;
        h[3] = (_Float16)v.w;
        o[i] = h;
    }
}

// ---------------------------------------------------------------------------
// WMMA GEMM: C[M,N] += A[M,K] * B[N,K]^T ; epilogue applies scale/bias.
// ---------------------------------------------------------------------------
__global__ __launch_bounds__(256)
void gemm_wmma_f16(const _Float16* __restrict__ A,   // [M, K] row-major
                   const _Float16* __restrict__ Bw,  // [N, K] row-major (= W^T access)
                   const float*    __restrict__ scale,
                   const float*    __restrict__ bias,
                   float*          __restrict__ Cout) {
    __shared__ __align__(16) _Float16 As[2][BM * LDS_STRIDE];   // 40 KB
    __shared__ __align__(16) _Float16 Bs[2][BN * LDS_STRIDE];   // 20 KB

    const int tid   = threadIdx.x;
    const int lane  = tid & 31;
    const int wave  = tid >> 5;
    const int waveM = wave & 3;          // 0..3 -> 64-row slab
    const int waveN = wave >> 2;         // 0..1 -> 64-col slab
    const int lrow  = lane & 15;
    const int hi    = lane >> 4;         // lane half selector

    const int mBlock = blockIdx.y * BM;
    const int nBlock = blockIdx.x * BN;

    // Cooperative global->LDS staging: chunk = 8 halves (16B).
    // A tile 256x32 -> 4 chunks/thread; B tile 128x32 -> 2 chunks/thread.
    const int r0 = tid >> 2;             // 0..63
    const int c0 = (tid & 3) * 8;
    long aOff[4], bOff[2];
#pragma unroll
    for (int i = 0; i < 4; ++i)
        aOff[i] = (long)(mBlock + r0 + i * 64) * IN_DIM + c0;
#pragma unroll
    for (int i = 0; i < 2; ++i)
        bOff[i] = (long)(nBlock + r0 + i * 64) * IN_DIM + c0;
    const int ldsOffA[4] = { r0 * LDS_STRIDE + c0,
                             (r0 +  64) * LDS_STRIDE + c0,
                             (r0 + 128) * LDS_STRIDE + c0,
                             (r0 + 192) * LDS_STRIDE + c0 };
    const int ldsOffB[2] = { r0 * LDS_STRIDE + c0,
                             (r0 +  64) * LDS_STRIDE + c0 };

    // f32 accumulators: 4 (M) x 4 (N) tiles of 16x16 = 128 VGPRs
    v8f acc[4][4];
    const v8f vzero = {0.f, 0.f, 0.f, 0.f, 0.f, 0.f, 0.f, 0.f};
#pragma unroll
    for (int mt = 0; mt < 4; ++mt)
#pragma unroll
        for (int nt = 0; nt < 4; ++nt)
            acc[mt][nt] = vzero;

    // Prologue: async-fill K-tile 0 into buffer 0
    {
#pragma unroll
        for (int i = 0; i < 4; ++i)
            async_copy_b128(lds_addr_of(&As[0][ldsOffA[i]]), A + aOff[i]);
#pragma unroll
        for (int i = 0; i < 2; ++i)
            async_copy_b128(lds_addr_of(&Bs[0][ldsOffB[i]]), Bw + bOff[i]);
        wait_asynccnt0();
    }
    __syncthreads();

    const int kTiles = IN_DIM / BK;      // 128
    int buf = 0;
    for (int kt = 0; kt < kTiles; ++kt) {
        const bool hasNext = (kt + 1) < kTiles;

        // Issue async fill of the other buffer (its readers all passed the
        // previous barrier), overlapping the DMA with this step's WMMAs.
        if (hasNext) {
            const long k = (long)(kt + 1) * BK;
            const int nb = buf ^ 1;
#pragma unroll
            for (int i = 0; i < 4; ++i)
                async_copy_b128(lds_addr_of(&As[nb][ldsOffA[i]]), A + aOff[i] + k);
#pragma unroll
            for (int i = 0; i < 2; ++i)
                async_copy_b128(lds_addr_of(&Bs[nb][ldsOffB[i]]), Bw + bOff[i] + k);
        }
        if (kt + 2 < kTiles) {   // GL2 warm-up for the +2 tile
            const long k2 = (long)(kt + 2) * BK;
            __builtin_prefetch(A  + aOff[0] + k2, 0, 0);
            __builtin_prefetch(Bw + bOff[0] + k2, 0, 0);
        }

        // --- B fragments (kept live across the mt loop) ---
        // B 32x16 f16 layout: lane<16 -> col N=lrow, K{0..7}+{16..23} ; lane>=16 -> K{8..15}+{24..31}
        v16h bfrag[4];
#pragma unroll
        for (int nt = 0; nt < 4; ++nt) {
            const _Float16* bp =
                &Bs[buf][(waveN * 64 + nt * 16 + lrow) * LDS_STRIDE + hi * 8];
            union { v16h v; v8h h[2]; } u;
            u.h[0] = *(const v8h*)(bp);
            u.h[1] = *(const v8h*)(bp + 16);
            bfrag[nt] = u.v;
        }

        // --- mt-major: one A fragment feeds 4 WMMAs (register recycling) ---
        // A 16x32 f16 layout: lane<16 -> row M=lrow, K{0..7}+{16..23};
        //                     lane>=16 -> same row,  K{8..15}+{24..31}
#pragma unroll
        for (int mt = 0; mt < 4; ++mt) {
            const _Float16* ap =
                &As[buf][(waveM * 64 + mt * 16 + lrow) * LDS_STRIDE + hi * 8];
            union { v16h v; v8h h[2]; } u;
            u.h[0] = *(const v8h*)(ap);
            u.h[1] = *(const v8h*)(ap + 16);
            const v16h afrag = u.v;
#pragma unroll
            for (int nt = 0; nt < 4; ++nt)
                acc[mt][nt] = __builtin_amdgcn_wmma_f32_16x16x32_f16(
                    /*neg_a=*/false, afrag,
                    /*neg_b=*/false, bfrag[nt],
                    /*c_mod=*/(short)0, acc[mt][nt],
                    /*reuse_a=*/false, /*reuse_b=*/false);
        }

        if (hasNext) {
            wait_asynccnt0();    // this wave's DMA chunks landed in LDS
            __syncthreads();     // all waves' chunks landed
            buf ^= 1;
        }
    }

    // Epilogue: C/D layout: VGPR r -> (M = r + hi*8, N = lrow within tile)
#pragma unroll
    for (int mt = 0; mt < 4; ++mt) {
#pragma unroll
        for (int nt = 0; nt < 4; ++nt) {
            const int col = nBlock + waveN * 64 + nt * 16 + lrow;
            const float sc = scale[col];
            const float bi = bias[col];
            const int rowBase = mBlock + waveM * 64 + mt * 16 + hi * 8;
            float* op = Cout + (long)rowBase * OUT_DIM + col;
#pragma unroll
            for (int r = 0; r < 8; ++r)
                op[(long)r * OUT_DIM] = acc[mt][nt][r] * sc + bi;
        }
    }
}

// ---------------------------------------------------------------------------
// Launch: d_in = { x[f32], weight_int8[i32], scale[f32], bias[f32] }
// workspace: w_f16 (32 MB) | x_f16 (64 MB)
// ---------------------------------------------------------------------------
extern "C" void kernel_launch(void* const* d_in, const int* in_sizes, int n_in,
                              void* d_out, int out_size, void* d_ws, size_t ws_size,
                              hipStream_t stream) {
    const float* x     = (const float*)d_in[0];
    const int*   w8    = (const int*)  d_in[1];
    const float* scale = (const float*)d_in[2];
    const float* bias  = (const float*)d_in[3];
    float*       out   = (float*)d_out;

    _Float16* wF = (_Float16*)d_ws;
    _Float16* xF = wF + (size_t)OUT_DIM * IN_DIM;

    const int n4w = (OUT_DIM * IN_DIM) / 4;
    convert_w_f16<<<(n4w + 255) / 256, 256, 0, stream>>>((const int4*)w8, (v4h*)wF, n4w);

    const int n4x = (M_DIM * IN_DIM) / 4;
    convert_x_f16<<<(n4x + 255) / 256, 256, 0, stream>>>((const float4*)x, (v4h*)xF, n4x);

    dim3 grid(OUT_DIM / BN, M_DIM / BM);   // 32 x 32 blocks
    gemm_wmma_f16<<<grid, 256, 0, stream>>>(xF, wF, scale, bias, out);
}
